// DeepGCN_953482740192
// MI455X (gfx1250) — matrix operand order repaired and checked by reading, not compiled, not run on hardware
//
#include <hip/hip_runtime.h>
#include <hip/hip_bf16.h>

typedef __attribute__((ext_vector_type(16))) __bf16 v16bf;
typedef __attribute__((ext_vector_type(8)))  __bf16 v8bf;
typedef __attribute__((ext_vector_type(8)))  float  v8f;
typedef __attribute__((ext_vector_type(4)))  float  v4f;

#define N_NODES 16384
#define NFEAT   512
#define NHID    128
#define NCLASS  8

// Build WMMA A-operand (16x32 bf16) for this lane from a row-major f32 row.
// Lane layout: lanes 0-15 -> M=lane, K = kb+0..7 and kb+16..23;
//              lanes 16-31 -> M=lane-16, K = k0+8..15 and k0+24..31.
// Caller passes kb = k0 + ((lane>>4)<<3).
// Loads are non-temporal: the A stream (adj / x) is single-use and must not
// evict the L2-resident bf16 B operands (XW / HWpad / weights).
static __device__ __forceinline__ v16bf load_a_f32_nt(const float* __restrict__ rowp, int kb) {
  v4f f0 = __builtin_nontemporal_load((const v4f*)(rowp + kb));
  v4f f1 = __builtin_nontemporal_load((const v4f*)(rowp + kb + 4));
  v4f f2 = __builtin_nontemporal_load((const v4f*)(rowp + kb + 16));
  v4f f3 = __builtin_nontemporal_load((const v4f*)(rowp + kb + 20));
  v16bf a;
#pragma unroll
  for (int j = 0; j < 4; ++j) {
    a[j]      = (__bf16)f0[j];
    a[4 + j]  = (__bf16)f1[j];
    a[8 + j]  = (__bf16)f2[j];
    a[12 + j] = (__bf16)f3[j];
  }
  return a;
}

// Weight prep: W1 f32[512,128] -> bf16 row-major; W2 f32[128,8] -> bf16 [128,16] zero-padded.
__global__ void convert_weights(const float* __restrict__ W1, const float* __restrict__ W2,
                                __bf16* __restrict__ W1b, __bf16* __restrict__ W2p) {
  int i = blockIdx.x * blockDim.x + threadIdx.x;
  if (i < NFEAT * NHID) W1b[i] = (__bf16)W1[i];
  int j = i - NFEAT * NHID;
  if (j >= 0 && j < NHID * 16) {
    int col = j & 15;
    W2p[j] = (col < NCLASS) ? (__bf16)W2[(j >> 4) * NCLASS + col] : (__bf16)0.0f;
  }
}

// C = [relu](A_f32[M,K] x B_bf16[K,128]) as bf16 row-major [M,128].
// Block: 256 threads = 8 waves; wave covers 16 rows x 64 cols (4 WMMA col-tiles).
// Block covers 64 rows x 128 cols; grid = M/64.
template <bool RELU>
__global__ __launch_bounds__(256) void gemm_a32(const float* __restrict__ A,
                                                const __bf16* __restrict__ B,
                                                __bf16* __restrict__ C,
                                                int K, int lda) {
  const int lane = threadIdx.x & 31;
  const int wave = threadIdx.x >> 5;
  const int m0   = blockIdx.x * 64 + (wave >> 1) * 16;
  const int n0   = (wave & 1) * 64;
  const int koff = (lane >> 4) << 3;
  const float* ap = A + (size_t)(m0 + (lane & 15)) * (size_t)lda;

  v8f acc[4];
#pragma unroll
  for (int t = 0; t < 4; ++t) acc[t] = (v8f){};

#pragma unroll 2
  for (int k0 = 0; k0 < K; k0 += 32) {
    // Unconditional speculative NT prefetch ~1 KiB ahead of this lane's A row
    // (speculative prefetch past the valid range is silently dropped).
    __builtin_prefetch(ap + k0 + koff + 256, 0, 0);
    v16bf a = load_a_f32_nt(ap, k0 + koff);
    const __bf16* bp = B + (size_t)(k0 + koff) * NHID + n0 + (lane & 15);  // B lane = N-col, elements = K (table order)
#pragma unroll
    for (int t = 0; t < 4; ++t) {
      v16bf b;
#pragma unroll
      for (int s = 0; s < 16; ++s) b[s] = bp[(size_t)((s & 7) + ((s >> 3) << 4)) * NHID + t * 16];
      acc[t] = __builtin_amdgcn_wmma_f32_16x16x32_bf16(false, a, false, b,
                                                       (short)0, acc[t], false, false);
    }
  }

  // C layout: lane=N, VGPR r = M (lanes>=16: M=r+8). Store row-major bf16.
  const int mrow = m0 + ((lane >> 4) << 3);
  const int ncol = n0 + (lane & 15);
#pragma unroll
  for (int t = 0; t < 4; ++t) {
#pragma unroll
    for (int r = 0; r < 8; ++r) {
      float v = acc[t][r];
      if (RELU) v = v > 0.0f ? v : 0.0f;
      C[(size_t)(mrow + r) * NHID + (ncol + t * 16)] = (__bf16)v;
    }
  }
}

// HWpad = H(bf16[M,128]) x W2pad(bf16[128,16]) -> bf16 [M,16]. Wave = 16 rows, 1 tile.
__global__ __launch_bounds__(256) void hw_kernel(const __bf16* __restrict__ H,
                                                 const __bf16* __restrict__ W2p,
                                                 __bf16* __restrict__ HW) {
  const int lane = threadIdx.x & 31;
  const int wave = threadIdx.x >> 5;
  const int m0   = blockIdx.x * 128 + wave * 16;
  const int koff = (lane >> 4) << 3;
  const __bf16* ap = H + (size_t)(m0 + (lane & 15)) * NHID;

  v8f acc = (v8f){};
#pragma unroll
  for (int k0 = 0; k0 < NHID; k0 += 32) {
    v8bf lo = *(const v8bf*)(ap + k0 + koff);
    v8bf hi = *(const v8bf*)(ap + k0 + koff + 16);
    v16bf a;
#pragma unroll
    for (int j = 0; j < 8; ++j) { a[j] = lo[j]; a[8 + j] = hi[j]; }
    v16bf b;
#pragma unroll
    for (int s = 0; s < 16; ++s) b[s] = W2p[(size_t)(k0 + koff + (s & 7) + ((s >> 3) << 4)) * 16 + (lane & 15)];
    acc = __builtin_amdgcn_wmma_f32_16x16x32_bf16(false, a, false, b,
                                                  (short)0, acc, false, false);
  }
  const int mrow = m0 + ((lane >> 4) << 3);
  const int ncol = lane & 15;
#pragma unroll
  for (int r = 0; r < 8; ++r)
    HW[(size_t)(mrow + r) * 16 + ncol] = (__bf16)acc[r];
}

// out = log_softmax(adj x HWpad + b2) over 8 real classes. Wave = 16 rows, 1 tile.
__global__ __launch_bounds__(256) void out_kernel(const float* __restrict__ adj,
                                                  const __bf16* __restrict__ HW,
                                                  const float* __restrict__ b2,
                                                  float* __restrict__ out) {
  const int lane = threadIdx.x & 31;
  const int wave = threadIdx.x >> 5;
  const int m0   = blockIdx.x * 128 + wave * 16;
  const int koff = (lane >> 4) << 3;
  const float* ap = adj + (size_t)(m0 + (lane & 15)) * (size_t)N_NODES;

  v8f acc = (v8f){};
#pragma unroll 2
  for (int k0 = 0; k0 < N_NODES; k0 += 32) {
    __builtin_prefetch(ap + k0 + koff + 256, 0, 0);
    v16bf a = load_a_f32_nt(ap, k0 + koff);
    v16bf b;  // cols 8..15 are zeros
#pragma unroll
    for (int s = 0; s < 16; ++s) b[s] = HW[(size_t)(k0 + koff + (s & 7) + ((s >> 3) << 4)) * 16 + (lane & 15)];
    acc = __builtin_amdgcn_wmma_f32_16x16x32_bf16(false, a, false, b,
                                                  (short)0, acc, false, false);
  }

  // Per VGPR r: lanes 0-7 = row (m0+r) cols 0-7; lanes 8-15 padding;
  // lanes 16-23 = row (m0+r+8) cols 0-7; lanes 24-31 padding.
  // xor-butterfly over {1,2,4} stays within each 8-lane group.
  const float bias  = b2[lane & 7];
  const int   mbase = m0 + ((lane >> 4) << 3);
#pragma unroll
  for (int r = 0; r < 8; ++r) {
    float v = acc[r] + bias;
    float mx = v;
#pragma unroll
    for (int off = 1; off < 8; off <<= 1) {
      float o = __shfl_xor(mx, off, 32);
      mx = o > mx ? o : mx;
    }
    float e = __expf(v - mx);
    float s = e;
#pragma unroll
    for (int off = 1; off < 8; off <<= 1) s += __shfl_xor(s, off, 32);
    float res = (v - mx) - __logf(s);
    if ((lane & 15) < 8)
      out[(size_t)(mbase + r) * NCLASS + (lane & 7)] = res;
  }
}

extern "C" void kernel_launch(void* const* d_in, const int* in_sizes, int n_in,
                              void* d_out, int out_size, void* d_ws, size_t ws_size,
                              hipStream_t stream) {
  const float* x   = (const float*)d_in[0];
  const float* adj = (const float*)d_in[1];
  const float* W1  = (const float*)d_in[2];
  const float* W2  = (const float*)d_in[3];
  const float* b2  = (const float*)d_in[4];
  float* out = (float*)d_out;

  // Workspace layout (all 256B-aligned offsets):
  char* ws = (char*)d_ws;
  __bf16* W1b = (__bf16*)(ws);                                   // 512*128*2   = 128 KiB
  __bf16* W2p = (__bf16*)(ws + (size_t)131072);                  // 128*16*2    =   4 KiB
  __bf16* XW  = (__bf16*)(ws + (size_t)131072 + 4096);           // 16384*128*2 =   4 MiB
  __bf16* H   = (__bf16*)(ws + (size_t)131072 + 4096
                             + (size_t)N_NODES * NHID * 2);      // 4 MiB
  __bf16* HW  = (__bf16*)(ws + (size_t)131072 + 4096
                             + (size_t)N_NODES * NHID * 4);      // 16384*16*2 = 512 KiB

  int conv_elems = NFEAT * NHID + NHID * 16;
  convert_weights<<<(conv_elems + 255) / 256, 256, 0, stream>>>(W1, W2, W1b, W2p);

  // XW = bf16(x @ W1)
  gemm_a32<false><<<N_NODES / 64, 256, 0, stream>>>(x, W1b, XW, NFEAT, NFEAT);
  // H = bf16(relu(adj @ XW))   -- 1 GiB adj stream #1 (NT)
  gemm_a32<true><<<N_NODES / 64, 256, 0, stream>>>(adj, XW, H, N_NODES, N_NODES);
  // HWpad = bf16(H @ W2pad)
  hw_kernel<<<N_NODES / 128, 256, 0, stream>>>(H, W2p, HW);
  // out = log_softmax(adj @ HWpad + b2)  -- 1 GiB adj stream #2 (NT)
  out_kernel<<<N_NODES / 128, 256, 0, stream>>>(adj, HW, b2, out);
}
